// GNN_10995116277976
// MI455X (gfx1250) — matrix operand order repaired and checked by reading, not compiled, not run on hardware
//
#include <hip/hip_runtime.h>
#include <hip/hip_bf16.h>

#define N_NODES 100000
#define N_EDGES 1600000
#define IN_DIM  1024
#define HID     64
#define OUT_DIM 1024
#define N_LAYERS 4
#define NET     196
#define MID_DIM 512
#define NEG_SLOPE 0.2f
#define BN_EPS  1e-5f

typedef float    v2f  __attribute__((ext_vector_type(2)));
typedef float    v8f  __attribute__((ext_vector_type(8)));
typedef _Float16 v16h __attribute__((ext_vector_type(16)));
typedef unsigned int v4u __attribute__((ext_vector_type(4)));

union V16H { v16h v; v4u q[2]; };

// ---------------- CSR build ----------------
__global__ void k_zero_i32(int* p, int n) {
  int i = blockIdx.x * blockDim.x + threadIdx.x;
  if (i < n) p[i] = 0;
}

__global__ void k_degree(const int* __restrict__ ei, int* __restrict__ deg) {
  int j = blockIdx.x * blockDim.x + threadIdx.x;           // exactly E threads
  atomicAdd(&deg[ei[N_EDGES + j]], 1);
}

__global__ void k_scan_block(const int* __restrict__ deg, int* __restrict__ off,
                             int* __restrict__ bsum) {
  __shared__ int s[256];
  int i = blockIdx.x * 256 + threadIdx.x;
  int v = (i < N_NODES) ? deg[i] : 0;
  s[threadIdx.x] = v;
  __syncthreads();
  for (int d = 1; d < 256; d <<= 1) {
    int t = (threadIdx.x >= d) ? s[threadIdx.x - d] : 0;
    __syncthreads();
    s[threadIdx.x] += t;
    __syncthreads();
  }
  if (i < N_NODES) off[i] = s[threadIdx.x] - v;            // exclusive within block
  if (threadIdx.x == 255) bsum[blockIdx.x] = s[255];
}

__global__ void k_scan_top(int* bsum, int nb) {
  __shared__ int s[512];
  int v = (threadIdx.x < nb) ? bsum[threadIdx.x] : 0;
  s[threadIdx.x] = v;
  __syncthreads();
  for (int d = 1; d < 512; d <<= 1) {
    int t = (threadIdx.x >= d) ? s[threadIdx.x - d] : 0;
    __syncthreads();
    s[threadIdx.x] += t;
    __syncthreads();
  }
  if (threadIdx.x < (unsigned)nb) bsum[threadIdx.x] = s[threadIdx.x] - v;  // exclusive
}

__global__ void k_scan_add(int* off, const int* __restrict__ bsum) {
  int i = blockIdx.x * 256 + threadIdx.x;
  if (i < N_NODES)       off[i] += bsum[i >> 8];
  else if (i == N_NODES) off[i]  = N_EDGES;
}

__global__ void k_scatter(const int* __restrict__ ei, const int* __restrict__ ea,
                          const int* __restrict__ off, int* __restrict__ cnt,
                          int* __restrict__ csrc, int* __restrict__ cattr) {
  int j = blockIdx.x * blockDim.x + threadIdx.x;           // exactly E threads
  int d = ei[N_EDGES + j];
  int pos = off[d] + atomicAdd(&cnt[d], 1);
  csrc[pos]  = ei[j];
  cattr[pos] = ea[j];
}

// ---------------- small precomputes ----------------
// tvals[l][net] = edge_emb[net] . (conv_We[l] @ att_edge[l])
__global__ void k_tvals(const float* __restrict__ We, const float* __restrict__ ae,
                        const float* __restrict__ emb, float* __restrict__ tv) {
  int t = blockIdx.x * blockDim.x + threadIdx.x;
  if (t >= N_LAYERS * NET) return;
  int l = t / NET, net = t % NET;
  const float* W = We + (size_t)l * HID * HID;
  const float* a = ae + (size_t)l * HID;
  const float* e = emb + (size_t)net * HID;
  float acc = 0.f;
  for (int k = 0; k < HID; ++k) {
    float wk = 0.f;
    for (int j = 0; j < HID; ++j) wk += W[k * HID + j] * a[j];
    acc += e[k] * wk;
  }
  tv[t] = acc;
}

// Pack row-major f32 [K x Ncol] into f16 WMMA-B layout:
// dst[(kg*Ncol + n)*16 + j] = src[(kg*16+j)*Ncol + n],  kg = k/16
__global__ void k_pack_b16(const float* __restrict__ src, _Float16* __restrict__ dst,
                           int K, int Ncol) {
  int id = blockIdx.x * blockDim.x + threadIdx.x;
  if (id >= (K / 16) * Ncol) return;
  int kg = id / Ncol, n = id % Ncol;
#pragma unroll
  for (int j = 0; j < 16; ++j)
    dst[(size_t)id * 16 + j] = (_Float16)src[(size_t)(kg * 16 + j) * Ncol + n];
}

// e output: e[j,:] = edge_emb[edge_attr[j],:]
__global__ void k_egather(const int* __restrict__ ea, const float* __restrict__ emb,
                          float* __restrict__ oute) {
  int g = blockIdx.x * blockDim.x + threadIdx.x;           // exactly E*16 threads
  int j = g >> 4, q = g & 15;
  const float4* s = (const float4*)(emb + (size_t)ea[j] * HID);
  ((float4*)(oute + (size_t)j * HID))[q] = s[q];
}

// ---------------- f32 WMMA GEMM: C[M x 64] = A[M x K] @ B[K x 64] (+bias) ----------------
__global__ __launch_bounds__(256) void k_gemm_f32(const float* __restrict__ A,
                                                  const float* __restrict__ B,
                                                  const float* __restrict__ bias,
                                                  float* __restrict__ C, int K) {
  const int lane   = threadIdx.x & 31;
  const int wave   = threadIdx.x >> 5;
  const int laneLo = lane & 15, laneHi = lane >> 4;
  const int r0 = blockIdx.x * 32 + (wave >> 2) * 16;       // 2 row-tiles x 4 col-tiles
  const int c0 = (wave & 3) * 16;
  const float* Arow = A + (size_t)(r0 + laneLo) * K + 2 * laneHi;
  const float* Bp   = B + c0 + laneLo + (size_t)(2 * laneHi) * HID;
  v8f acc = {};
#pragma unroll 4
  for (int k = 0; k < K; k += 4) {
    v2f a; a.x = Arow[k]; a.y = Arow[k + 1];
    v2f b; b.x = Bp[(size_t)k * HID]; b.y = Bp[(size_t)k * HID + HID];
    acc = __builtin_amdgcn_wmma_f32_16x16x4_f32(false, a, false, b, (short)0, acc,
                                                false, false);
  }
  float bb = bias ? bias[c0 + laneLo] : 0.f;
#pragma unroll
  for (int r = 0; r < 8; ++r) {
    int row = r0 + r + laneHi * 8;
    C[(size_t)row * HID + c0 + laneLo] = acc[r] + bb;
  }
}

// ---------------- per-node attention scalars ----------------
__global__ void k_att(const float* __restrict__ hh, const float* __restrict__ asv,
                      const float* __restrict__ adv, float* __restrict__ as_out,
                      float* __restrict__ ad_out) {
  int n = blockIdx.x * blockDim.x + threadIdx.x;
  if (n >= N_NODES) return;
  const float4* hr = (const float4*)(hh + (size_t)n * HID);
  const float4* s  = (const float4*)asv;
  const float4* d  = (const float4*)adv;
  float accs = 0.f, accd = 0.f;
#pragma unroll
  for (int q = 0; q < 16; ++q) {
    float4 h4 = hr[q], s4 = s[q], d4 = d[q];
    accs += h4.x * s4.x + h4.y * s4.y + h4.z * s4.z + h4.w * s4.w;
    accd += h4.x * d4.x + h4.y * d4.y + h4.z * d4.z + h4.w * d4.w;
  }
  as_out[n] = accs;
  ad_out[n] = accd;
}

// ---------------- fused online-softmax aggregation + bias/BN/ReLU/residual ----------------
// one wave per node; lane holds 2 columns
__global__ __launch_bounds__(256) void k_agg(
    const int* __restrict__ off, const int* __restrict__ csrc,
    const int* __restrict__ cattr, const float* __restrict__ tv,
    const float* __restrict__ a_src, const float* __restrict__ a_dst,
    const float* __restrict__ hh, const float* __restrict__ cb,
    const float* __restrict__ bg, const float* __restrict__ bb,
    const float* __restrict__ bm, const float* __restrict__ bv,
    float* __restrict__ h) {
  const int lane = threadIdx.x & 31;
  const int n    = blockIdx.x * 8 + (threadIdx.x >> 5);    // N/8 blocks exactly
  const float adn = a_dst[n];
  const int beg = off[n], end = off[n + 1];
  float m = -__builtin_inff(), z = 0.f, ax = 0.f, ay = 0.f;
  for (int e = beg; e < end; ++e) {
    int s = csrc[e];
    float a = a_src[s] + adn + tv[cattr[e]];
    a = (a > 0.f) ? a : NEG_SLOPE * a;
    const float2 hv = *(const float2*)(hh + (size_t)s * HID + lane * 2);
    if (a > m) {
      float sc = (m == -__builtin_inff()) ? 0.f : __expf(m - a);
      z  = z  * sc + 1.f;
      ax = ax * sc + hv.x;
      ay = ay * sc + hv.y;
      m = a;
    } else {
      float w = __expf(a - m);
      z += w; ax += w * hv.x; ay += w * hv.y;
    }
  }
  const float inv = 1.f / (z + 1e-16f);
  const int c = lane * 2;
  float2 cbv = *(const float2*)(cb + c);
  float2 bgv = *(const float2*)(bg + c);
  float2 bbv = *(const float2*)(bb + c);
  float2 bmv = *(const float2*)(bm + c);
  float2 bvv = *(const float2*)(bv + c);
  float2 hold = *(const float2*)(h + (size_t)n * HID + c);
  float sx = bgv.x * rsqrtf(bvv.x + BN_EPS);
  float sy = bgv.y * rsqrtf(bvv.y + BN_EPS);
  float ox = (ax * inv + cbv.x - bmv.x) * sx + bbv.x;
  float oy = (ay * inv + cbv.y - bmv.y) * sy + bbv.y;
  ox = fmaxf(ox, 0.f); oy = fmaxf(oy, 0.f);
  float2 r; r.x = ox + hold.x; r.y = oy + hold.y;
  *(float2*)(h + (size_t)n * HID + c) = r;
}

// ---------------- fused readout: y = sigmoid(h@W1+b1)@W2+b2, f16 WMMA ----------------
__global__ __launch_bounds__(256) void k_readout(
    const float* __restrict__ h, const _Float16* __restrict__ w1h,
    const float* __restrict__ b1, const _Float16* __restrict__ w2h,
    const float* __restrict__ b2, float* __restrict__ y) {
  __shared__ __align__(16) _Float16 h16[16 * HID];
  __shared__ __align__(16) _Float16 mid[16 * MID_DIM];
  const int r0 = blockIdx.x * 16;
  {
    int base = threadIdx.x * 4;                            // 256 threads * 4 = 16*64
    const float4 v = *(const float4*)(h + (size_t)r0 * HID + base);
    h16[base + 0] = (_Float16)v.x; h16[base + 1] = (_Float16)v.y;
    h16[base + 2] = (_Float16)v.z; h16[base + 3] = (_Float16)v.w;
  }
  __syncthreads();
  const int lane   = threadIdx.x & 31;
  const int wave   = threadIdx.x >> 5;
  const int laneLo = lane & 15, laneHi = lane >> 4;

  // Stage 1: mid[16 x 512], this wave covers cols [wave*64, wave*64+64)
  for (int t = 0; t < 4; ++t) {
    const int c0 = wave * 64 + t * 16;
    v8f acc = {};
#pragma unroll
    for (int kt = 0; kt < HID / 32; ++kt) {
      V16H a, b;
      const _Float16* ap = &h16[laneLo * HID + kt * 32 + laneHi * 16];
      a.q[0] = *(const v4u*)ap;
      a.q[1] = *(const v4u*)(ap + 8);
      const _Float16* bp = w1h + ((size_t)(kt * 2 + laneHi) * MID_DIM + c0 + laneLo) * 16;
      b.q[0] = *(const v4u*)bp;
      b.q[1] = *(const v4u*)(bp + 8);
      acc = __builtin_amdgcn_wmma_f32_16x16x32_f16(false, a.v, false, b.v, (short)0,
                                                   acc, false, false);
    }
    float bc = b1[c0 + laneLo];
#pragma unroll
    for (int r = 0; r < 8; ++r) {
      float x  = acc[r] + bc;
      float sg = 1.f / (1.f + __expf(-x));
      mid[(r + laneHi * 8) * MID_DIM + c0 + laneLo] = (_Float16)sg;
    }
  }
  __syncthreads();

  // Stage 2: y[16 x 1024], this wave covers cols [wave*128, wave*128+128)
  for (int t = 0; t < 8; ++t) {
    const int c0 = wave * 128 + t * 16;
    v8f acc = {};
#pragma unroll 4
    for (int kt = 0; kt < MID_DIM / 32; ++kt) {
      V16H a, b;
      const _Float16* ap = &mid[laneLo * MID_DIM + kt * 32 + laneHi * 16];
      a.q[0] = *(const v4u*)ap;
      a.q[1] = *(const v4u*)(ap + 8);
      const _Float16* bp = w2h + ((size_t)(kt * 2 + laneHi) * OUT_DIM + c0 + laneLo) * 16;
      b.q[0] = *(const v4u*)bp;
      b.q[1] = *(const v4u*)(bp + 8);
      acc = __builtin_amdgcn_wmma_f32_16x16x32_f16(false, a.v, false, b.v, (short)0,
                                                   acc, false, false);
    }
    float bc = b2[c0 + laneLo];
#pragma unroll
    for (int r = 0; r < 8; ++r)
      y[(size_t)(r0 + r + laneHi * 8) * OUT_DIM + c0 + laneLo] = acc[r] + bc;
  }
}

// ---------------- host launch ----------------
extern "C" void kernel_launch(void* const* d_in, const int* in_sizes, int n_in,
                              void* d_out, int out_size, void* d_ws, size_t ws_size,
                              hipStream_t stream) {
  const float* x        = (const float*)d_in[0];
  const int*   ei       = (const int*)d_in[1];
  const int*   eattr    = (const int*)d_in[2];
  const float* node_W   = (const float*)d_in[3];
  const float* node_b   = (const float*)d_in[4];
  const float* edge_emb = (const float*)d_in[5];
  const float* conv_W   = (const float*)d_in[6];
  const float* conv_We  = (const float*)d_in[7];
  const float* att_src  = (const float*)d_in[8];
  const float* att_dst  = (const float*)d_in[9];
  const float* att_edge = (const float*)d_in[10];
  const float* conv_b   = (const float*)d_in[11];
  const float* bn_g     = (const float*)d_in[12];
  const float* bn_b     = (const float*)d_in[13];
  const float* bn_m     = (const float*)d_in[14];
  const float* bn_v     = (const float*)d_in[15];
  const float* ro_W1    = (const float*)d_in[16];
  const float* ro_b1    = (const float*)d_in[17];
  const float* ro_W2    = (const float*)d_in[18];
  const float* ro_b2    = (const float*)d_in[19];

  float* y    = (float*)d_out;
  float* oute = y + (size_t)N_NODES * OUT_DIM;

  char* ws = (char*)d_ws;
  size_t o = 0;
  auto alloc = [&](size_t bytes) -> char* {
    char* p = ws + o;
    o = (o + bytes + 255) & ~(size_t)255;
    return p;
  };
  float*    h    = (float*)alloc((size_t)N_NODES * HID * 4);
  float*    hh   = (float*)alloc((size_t)N_NODES * HID * 4);
  float*    a_s  = (float*)alloc((size_t)N_NODES * 4);
  float*    a_d  = (float*)alloc((size_t)N_NODES * 4);
  int*      deg  = (int*)alloc((size_t)N_NODES * 4);
  int*      off  = (int*)alloc((size_t)(N_NODES + 1) * 4);
  int*      bsum = (int*)alloc(512 * 4);
  int*      csrc = (int*)alloc((size_t)N_EDGES * 4);
  int*      catt = (int*)alloc((size_t)N_EDGES * 4);
  float*    tv   = (float*)alloc((size_t)N_LAYERS * NET * 4);
  _Float16* w1h  = (_Float16*)alloc((size_t)HID * MID_DIM * 2);
  _Float16* w2h  = (_Float16*)alloc((size_t)MID_DIM * OUT_DIM * 2);

  const int nb = (N_NODES + 255) / 256;  // 391

  // CSR build (by dst)
  k_zero_i32<<<nb, 256, 0, stream>>>(deg, N_NODES);
  k_degree<<<N_EDGES / 256, 256, 0, stream>>>(ei, deg);
  k_scan_block<<<nb, 256, 0, stream>>>(deg, off, bsum);
  k_scan_top<<<1, 512, 0, stream>>>(bsum, nb);
  k_scan_add<<<nb, 256, 0, stream>>>(off, bsum);
  k_zero_i32<<<nb, 256, 0, stream>>>(deg, N_NODES);
  k_scatter<<<N_EDGES / 256, 256, 0, stream>>>(ei, eattr, off, deg, csrc, catt);

  // small precomputes
  k_tvals<<<(N_LAYERS * NET + 255) / 256, 256, 0, stream>>>(conv_We, att_edge,
                                                            edge_emb, tv);
  k_pack_b16<<<((HID / 16) * MID_DIM + 255) / 256, 256, 0, stream>>>(ro_W1, w1h,
                                                                     HID, MID_DIM);
  k_pack_b16<<<((MID_DIM / 16) * OUT_DIM + 255) / 256, 256, 0, stream>>>(ro_W2, w2h,
                                                                         MID_DIM, OUT_DIM);
  // e output (second tuple element)
  k_egather<<<(N_EDGES * 16) / 256, 256, 0, stream>>>(eattr, edge_emb, oute);

  // h = x @ node_W + node_b  (f32 WMMA, K=1024)
  k_gemm_f32<<<N_NODES / 32, 256, 0, stream>>>(x, node_W, node_b, h, IN_DIM);

  // layers
  for (int l = 0; l < N_LAYERS; ++l) {
    k_gemm_f32<<<N_NODES / 32, 256, 0, stream>>>(h, conv_W + (size_t)l * HID * HID,
                                                 (const float*)nullptr, hh, HID);
    k_att<<<nb, 256, 0, stream>>>(hh, att_src + (size_t)l * HID,
                                  att_dst + (size_t)l * HID, a_s, a_d);
    k_agg<<<N_NODES / 8, 256, 0, stream>>>(off, csrc, catt, tv + (size_t)l * NET,
                                           a_s, a_d, hh,
                                           conv_b + (size_t)l * HID,
                                           bn_g + (size_t)l * HID,
                                           bn_b + (size_t)l * HID,
                                           bn_m + (size_t)l * HID,
                                           bn_v + (size_t)l * HID, h);
  }

  // fused readout (f16 WMMA both GEMMs)
  k_readout<<<N_NODES / 16, 256, 0, stream>>>(h, w1h, ro_b1, w2h, ro_b2, y);

  (void)in_sizes; (void)n_in; (void)out_size; (void)ws_size;
}
